// projector_t_48627619725570
// MI455X (gfx1250) — matrix operand order repaired and checked by reading, not compiled, not run on hardware
//
#include <hip/hip_runtime.h>

// ---------------- fixed CT geometry ----------------
#define IMGSZ   512
#define NDET    768
#define NVIEWS  360
#define NBATCH  2
#define NROWS   (NBATCH * NVIEWS)        // 720 sinogram rows

constexpr float PIXF   = 0.7433f;
constexpr float DSOF   = 595.0f;
constexpr float DSDF   = 595.0f + 490.6f;
constexpr float DETF   = 1.2858f;
constexpr float DGAMMA = DETF / DSDF;    // equiangular pitch (rad)
constexpr float PIF    = 3.14159265358979323846f;

// ---------------- WMMA types (CDNA5, wave32) ----------------
typedef __attribute__((ext_vector_type(16))) __bf16 v16bf;
typedef __attribute__((ext_vector_type(8)))  float  v8f;

union FragBF16 {                 // 8 dwords == 16 bf16 == one A/B fragment
    unsigned int u[8];
    v16bf        v;
};

// ---------------- bf16 split helpers ----------------
__device__ inline unsigned short f32_to_bf16_rn(float x) {
    unsigned int u = __float_as_uint(x);
    unsigned int r = u + 0x7FFFu + ((u >> 16) & 1u);   // round-to-nearest-even
    return (unsigned short)(r >> 16);
}
__device__ inline float bf16_to_f32(unsigned short h) {
    return __uint_as_float(((unsigned int)h) << 16);
}
__device__ inline void split_bf16(float x, unsigned short* hi, unsigned short* lo) {
    unsigned short h = f32_to_bf16_rn(x);
    *hi = h;
    *lo = f32_to_bf16_rn(x - bf16_to_f32(h));
}

// =====================================================================
// Kernel 1: build Toeplitz ramp matrix, column-major for B-fragments:
//   Kb[j][k] = g(j - k) * DGAMMA   (j = output det, k = input det)
// stored as bf16 hi/lo split, row length NDET per j.
// =====================================================================
__global__ __launch_bounds__(256) void build_ramp_kernel(unsigned short* __restrict__ Kb_hi,
                                                         unsigned short* __restrict__ Kb_lo) {
    int idx = blockIdx.x * 256 + threadIdx.x;          // NDET*NDET total
    if (idx >= NDET * NDET) return;
    int j = idx / NDET;
    int k = idx - j * NDET;
    int n = j - k;
    float g;
    if (n == 0) {
        g = 1.0f / (8.0f * DGAMMA * DGAMMA);
    } else if (n & 1) {
        float s = __sinf((float)n * DGAMMA);
        g = -0.5f / (PIF * PIF * s * s);
    } else {
        g = 0.0f;
    }
    float val = g * DGAMMA;                            // fold trailing *DGAMMA of filter_proj
    unsigned short hi, lo;
    split_bf16(val, &hi, &lo);
    Kb_hi[idx] = hi;
    Kb_lo[idx] = lo;
}

// =====================================================================
// Kernel 2: cosine weighting + bf16 split of sinogram rows
//   p1[r][k] = proj[r][k] * DSO * cos(gamma_k)
// =====================================================================
__global__ __launch_bounds__(256) void weight_split_kernel(const float* __restrict__ proj,
                                                           unsigned short* __restrict__ A_hi,
                                                           unsigned short* __restrict__ A_lo) {
    int idx = blockIdx.x * 256 + threadIdx.x;          // NROWS*NDET total
    if (idx >= NROWS * NDET) return;
    int k = idx % NDET;
    float gam = ((float)k - (float)(NDET - 1) * 0.5f) * DGAMMA;
    float p = proj[idx] * (DSOF * __cosf(gam));
    unsigned short hi, lo;
    split_bf16(p, &hi, &lo);
    A_hi[idx] = hi;
    A_lo[idx] = lo;
}

// =====================================================================
// Kernel 3: ramp filtering as GEMM with bf16 WMMA, split-precision:
//   q[720 x 768] = A[720 x 768] * K[768 x 768]
// One wave (32 threads) per 16x16 output tile, K swept 32 at a time.
// Fragments follow ISA 7.12.2 layouts:
//   A (16x32 bf16): lane half selects K-subblock; K packed 2/dword.
//   B (32x16 bf16): lane = column, K contiguous 2/dword (Kb stored col-major).
// =====================================================================
__global__ __launch_bounds__(32) void filter_gemm_kernel(const unsigned short* __restrict__ A_hi,
                                                         const unsigned short* __restrict__ A_lo,
                                                         const unsigned short* __restrict__ Kb_hi,
                                                         const unsigned short* __restrict__ Kb_lo,
                                                         float* __restrict__ q) {
    const int lane = threadIdx.x;                      // 0..31 (wave32)
    const int tm = blockIdx.y;                         // 45 row tiles (720/16)
    const int tn = blockIdx.x;                         // 48 col tiles (768/16)
    const int l15 = lane & 15;
    const int hw  = lane >> 4;

    const int row = tm * 16 + l15;                     // A row (M = lane&15 in both halves)
    const int col = tn * 16 + l15;                     // B column (N = lane&15)

    const unsigned int* Ah32 = (const unsigned int*)A_hi;
    const unsigned int* Al32 = (const unsigned int*)A_lo;
    const unsigned int* Bh32 = (const unsigned int*)Kb_hi;
    const unsigned int* Bl32 = (const unsigned int*)Kb_lo;
    const int rowBase = row * (NDET / 2);              // dword index of A row
    const int colBase = col * (NDET / 2);              // dword index of Kb row (col-major K)

    v8f c = {0.f, 0.f, 0.f, 0.f, 0.f, 0.f, 0.f, 0.f};

    for (int kb = 0; kb < NDET / 32; ++kb) {           // 24 K-steps
        const int kbase = kb * 32;
        FragBF16 ah, al, bh, bl;
#pragma unroll
        for (int v = 0; v < 8; ++v) {
            // A: elements e=2v,2v+1 -> K = kbase + (e>>3)*16 + hw*8 + (e&7)
            int ak = kbase + ((v < 4) ? (hw * 8 + 2 * v) : (8 + hw * 8 + 2 * v));
            ah.u[v] = Ah32[rowBase + (ak >> 1)];
            al.u[v] = Al32[rowBase + (ak >> 1)];
            // B: elements e=2v,2v+1 -> K = kbase + (e>>3)*16 + hw*8 + (e&7)
            int bk = kbase + ((v < 4) ? (hw * 8 + 2 * v) : (8 + hw * 8 + 2 * v));
            bh.u[v] = Bh32[colBase + (bk >> 1)];
            bl.u[v] = Bl32[colBase + (bk >> 1)];
        }
        // split-precision accumulate: hi*hi + hi*lo + lo*hi (f32 accumulator)
        c = __builtin_amdgcn_wmma_f32_16x16x32_bf16(false, ah.v, false, bh.v,
                                                    (short)0, c, false, false);
        c = __builtin_amdgcn_wmma_f32_16x16x32_bf16(false, ah.v, false, bl.v,
                                                    (short)0, c, false, false);
        c = __builtin_amdgcn_wmma_f32_16x16x32_bf16(false, al.v, false, bh.v,
                                                    (short)0, c, false, false);
    }

    // C/D layout: VGPR r -> M = r + hw*8, N = lane&15
#pragma unroll
    for (int v = 0; v < 8; ++v) {
        int m = tm * 16 + v + hw * 8;
        q[m * NDET + col] = c[v];
    }
}

// =====================================================================
// Kernel 4: pixel-driven equiangular fan-beam backprojection.
// 16x16 pixel tile per 256-thread block (8 waves). Per view, both batch
// rows of q are staged in LDS; cos/sin(beta) table precomputed in LDS.
// dot > 0 inside FOV (DSO=595 >> max |pixel| ~ 269mm) so atan2 -> atanf.
// =====================================================================
__global__ __launch_bounds__(256) void backproject_kernel(const float* __restrict__ q,
                                                          float* __restrict__ out) {
    __shared__ float qv[NBATCH][NDET];
    __shared__ float cbs[NVIEWS];
    __shared__ float sbs[NVIEWS];

    const int t = threadIdx.x;
    for (int v = t; v < NVIEWS; v += 256) {
        float b = (float)v * (2.0f * PIF / (float)NVIEWS);
        cbs[v] = __cosf(b);
        sbs[v] = __sinf(b);
    }

    const int px = blockIdx.x * 16 + (t & 15);
    const int py = blockIdx.y * 16 + (t >> 4);
    const float X = ((float)px - (float)(IMGSZ - 1) * 0.5f) * PIXF;
    const float Y = ((float)py - (float)(IMGSZ - 1) * 0.5f) * PIXF;
    const float invdg = 1.0f / DGAMMA;
    const float ctr = (float)(NDET - 1) * 0.5f;

    float acc0 = 0.0f, acc1 = 0.0f;

    for (int v = 0; v < NVIEWS; ++v) {
        __syncthreads();                               // also fences cbs/sbs init at v=0
        for (int d = t; d < NDET; d += 256) {
            qv[0][d] = q[v * NDET + d];                // batch 0 row
            qv[1][d] = q[(NVIEWS + v) * NDET + d];     // batch 1 row
        }
        __syncthreads();
        if (v + 1 < NVIEWS)                            // gfx1250 global_prefetch_b8
            __builtin_prefetch(&q[(v + 1) * NDET + ((t & 63) << 2)], 0, 1);

        const float cb = cbs[v], sb = sbs[v];
        const float vx = X - DSOF * cb;
        const float vy = Y - DSOF * sb;
        const float dotc  = -(cb * vx + sb * vy);      // = DSO - (cb*X + sb*Y) > 0
        const float cross = sb * X - cb * Y;           // = -(cb*vy - sb*vx)
        const float g = atanf(cross / dotc) * invdg + ctr;

        const bool inb = (g >= 0.0f) && (g <= (float)(NDET - 1));
        int i0 = (int)floorf(g);
        i0 = (i0 < 0) ? 0 : ((i0 > NDET - 2) ? NDET - 2 : i0);
        float w = g - (float)i0;
        w = fminf(fmaxf(w, 0.0f), 1.0f);
        const float w1 = 1.0f - w;

        const float L2  = vx * vx + vy * vy;
        const float rL2 = __builtin_amdgcn_rcpf(L2);

        if (inb) {
            acc0 += (qv[0][i0] * w1 + qv[0][i0 + 1] * w) * rL2;
            acc1 += (qv[1][i0] * w1 + qv[1][i0 + 1] * w) * rL2;
        }
    }

    const float scale = 2.0f * PIF / (float)NVIEWS;
    out[py * IMGSZ + px]                   = acc0 * scale;
    out[IMGSZ * IMGSZ + py * IMGSZ + px]   = acc1 * scale;
}

// =====================================================================
// Host-side launch
// =====================================================================
extern "C" void kernel_launch(void* const* d_in, const int* in_sizes, int n_in,
                              void* d_out, int out_size, void* d_ws, size_t ws_size,
                              hipStream_t stream) {
    (void)in_sizes; (void)n_in; (void)out_size; (void)ws_size;
    const float* proj = (const float*)d_in[0];         // [2,360,768] f32
    float* out = (float*)d_out;                        // [2,512,512] f32

    // workspace layout (bytes), 256B-aligned sections (~6.5 MB total)
    char* ws = (char*)d_ws;
    unsigned short* Kb_hi = (unsigned short*)(ws + 0);                 // 768*768*2
    unsigned short* Kb_lo = (unsigned short*)(ws + 1179648);           // 768*768*2
    unsigned short* A_hi  = (unsigned short*)(ws + 2359296);           // 720*768*2
    unsigned short* A_lo  = (unsigned short*)(ws + 3465216);           // 720*768*2
    float*          q     = (float*)        (ws + 4571136);            // 720*768*4

    build_ramp_kernel<<<(NDET * NDET + 255) / 256, 256, 0, stream>>>(Kb_hi, Kb_lo);
    weight_split_kernel<<<(NROWS * NDET + 255) / 256, 256, 0, stream>>>(proj, A_hi, A_lo);

    dim3 ggrid(NDET / 16, NROWS / 16);                 // 48 x 45 tiles
    filter_gemm_kernel<<<ggrid, 32, 0, stream>>>(A_hi, A_lo, Kb_hi, Kb_lo, q);

    dim3 bgrid(IMGSZ / 16, IMGSZ / 16);                // 32 x 32 pixel tiles
    backproject_kernel<<<bgrid, 256, 0, stream>>>(q, out);
}
